// Wordwise_90958817395456
// MI455X (gfx1250) — matrix operand order repaired and checked by reading, not compiled, not run on hardware
//
#include <hip/hip_runtime.h>

typedef __attribute__((ext_vector_type(16))) __bf16 v16bf;
typedef __attribute__((ext_vector_type(8)))  float  v8f;
typedef __attribute__((ext_vector_type(4)))  int    v4i;

#define NBATCH 16
#define TFRAMES 4096
#define NWORDS 256
#define HCH 128

#if defined(__gfx1250__) && __has_builtin(__builtin_amdgcn_global_load_async_to_lds_b128)
#define ASYNC_STAGE 1
#else
#define ASYNC_STAGE 0
#endif

// ---------------------------------------------------------------------------
// Transpose features [B][80][T] f32 -> [B][T][96] bf16 (zero-pad ch 80..95)
// ---------------------------------------------------------------------------
__global__ void transpose_in_kernel(const float* __restrict__ x, __bf16* __restrict__ out) {
    size_t i = (size_t)blockIdx.x * blockDim.x + threadIdx.x;
    size_t total = (size_t)NBATCH * TFRAMES * 96;
    if (i >= total) return;
    int ci = (int)(i % 96);
    size_t bt = i / 96;
    int t = (int)(bt % TFRAMES);
    int b = (int)(bt / TFRAMES);
    float v = (ci < 80) ? x[((size_t)b * 80 + ci) * TFRAMES + t] : 0.f;
    out[i] = (__bf16)v;
}

// ---------------------------------------------------------------------------
// Pack conv weights [128][CIN][5] f32 into WMMA A-fragment order (bf16):
// out[ct][c][lane][j] where ct = Cout/16 tile, c = 32-wide K chunk,
// K flattening: kflat = k*CIN_P + ci ; A layout per CDNA5 ISA 16-bit 16x32.
// ---------------------------------------------------------------------------
__global__ void pack_weights_kernel(const float* __restrict__ w, __bf16* __restrict__ out,
                                    int CIN, int CIN_P) {
    int NCH = CIN_P * 5 / 32;
    int total = 8 * NCH * 512;
    int i = blockIdx.x * blockDim.x + threadIdx.x;
    if (i >= total) return;
    int j    = i & 15;
    int lane = (i >> 4) & 31;
    int c    = (i >> 9) % NCH;
    int ct   = i / (NCH * 512);
    int hi   = lane >> 4;
    int m    = ct * 16 + (lane & 15);
    int kk   = ((j < 8) ? 0 : 16) + hi * 8 + (j & 7);   // K index within chunk
    int kflat = c * 32 + kk;
    int k  = kflat / CIN_P;
    int ci = kflat % CIN_P;
    float v = (ci < CIN) ? w[((size_t)m * CIN + ci) * 5 + k] : 0.f;
    out[i] = (__bf16)v;
}

// ---------------------------------------------------------------------------
// Implicit-GEMM conv1d (K=5, 'same'): x bf16 [B][T][CIN_P] -> y bf16 [B][T][128]
// One block: 8 waves, each wave = one 16-row Cout tile over a 64-wide time tile.
// Input tile staged with GLOBAL_LOAD_ASYNC_TO_LDS_B128 (ASYNCcnt) when available.
// ---------------------------------------------------------------------------
template <int CIN_P, bool RELU>
__global__ __launch_bounds__(256) void conv_wmma_kernel(
    const __bf16* __restrict__ x, const __bf16* __restrict__ wpk,
    const float* __restrict__ bias, __bf16* __restrict__ y, int T) {
    constexpr int NT = 64;
    constexpr int NCH = CIN_P * 5 / 32;
    constexpr int XSPAN = NT + 4;
    constexpr int LSTR = CIN_P + 8;       // LDS row stride (bank-conflict pad)
    constexpr int ROWV = CIN_P / 8;       // uint4 per activation row
    __shared__ __bf16 xs[XSPAN * LSTR];

    const int b = blockIdx.y;
    const int t0 = blockIdx.x * NT;
    const int tid = threadIdx.x;
    const int lane = tid & 31;
    const int wave = tid >> 5;
    const int hi = lane >> 4;
    const int nl = lane & 15;

    // Stage [t0-2, t0+NT+2) rows into LDS (zeros outside [0,T))
    const uint4* xg = (const uint4*)x + (size_t)b * T * ROWV;
    for (int i = tid; i < XSPAN * ROWV; i += 256) {
        int tl = i / ROWV;
        int cv = i - tl * ROWV;
        int tt = t0 - 2 + tl;
        __bf16* ldsp = &xs[tl * LSTR + cv * 8];
        if (tt >= 0 && tt < T) {
#if ASYNC_STAGE
            __builtin_amdgcn_global_load_async_to_lds_b128(
                (__attribute__((address_space(1))) v4i*)(xg + (size_t)tt * ROWV + cv),
                (__attribute__((address_space(3))) v4i*)ldsp, 0, 0);
#else
            *(uint4*)ldsp = xg[(size_t)tt * ROWV + cv];
#endif
        } else {
            uint4 z = {0u, 0u, 0u, 0u};
            *(uint4*)ldsp = z;
        }
    }

    // Register-resident A fragments (pre-packed), reused for all 4 N-subtiles.
    // These global loads overlap the async LDS staging above.
    union V16 { v16bf v; uint4 q[2]; };
    V16 a[NCH];
    const uint4* wq = (const uint4*)wpk + (size_t)wave * NCH * 64;  // 512 bf16 = 64 uint4 / chunk
    #pragma unroll
    for (int c = 0; c < NCH; ++c) {
        a[c].q[0] = wq[c * 64 + lane * 2 + 0];
        a[c].q[1] = wq[c * 64 + lane * 2 + 1];
    }

    const int co0 = wave * 16 + hi * 8;   // this lane's 8 output channels
    float bv[8];
    #pragma unroll
    for (int r = 0; r < 8; ++r) bv[r] = bias[co0 + r];

#if ASYNC_STAGE
#if __has_builtin(__builtin_amdgcn_s_wait_asynccnt)
    __builtin_amdgcn_s_wait_asynccnt(0);
#else
    asm volatile("s_wait_asynccnt 0x0" ::: "memory");
#endif
#endif
    __syncthreads();

    #pragma unroll
    for (int n0 = 0; n0 < NT; n0 += 16) {
        v8f acc = {0.f, 0.f, 0.f, 0.f, 0.f, 0.f, 0.f, 0.f};
        #pragma unroll
        for (int c = 0; c < NCH; ++c) {
            const int k   = (c * 32) / CIN_P;   // tap (constant within chunk)
            const int ci0 = (c * 32) % CIN_P;
            V16 bf;
            const __bf16* src = &xs[(n0 + nl + k) * LSTR + ci0 + hi * 8];
            bf.q[0] = *(const uint4*)src;
            bf.q[1] = *(const uint4*)(src + 16);
            acc = __builtin_amdgcn_wmma_f32_16x16x32_bf16(
                false, a[c].v, false, bf.v, (short)0, acc, false, false);
        }
        const int tout = t0 + n0 + nl;
        union { __bf16 h[8]; uint4 u; } o;
        #pragma unroll
        for (int r = 0; r < 8; ++r) {
            float v = acc[r] + bv[r];
            if (RELU) v = fmaxf(v, 0.f);
            o.h[r] = (__bf16)v;
        }
        *(uint4*)&y[((size_t)b * T + tout) * HCH + co0] = o.u;
    }
}

// ---------------------------------------------------------------------------
// Segment-mean pool: x bf16 [B][T][128] -> out bf16 [B][W][128]
// ---------------------------------------------------------------------------
__global__ __launch_bounds__(128) void pool_kernel(
    const __bf16* __restrict__ x, const int* __restrict__ bounds,
    const int* __restrict__ lens, __bf16* __restrict__ out) {
    int wd = blockIdx.x;
    int b = blockIdx.y;
    int ci = threadIdx.x;
    int start = bounds[((size_t)b * 2 + 0) * NWORDS + wd];
    int end   = bounds[((size_t)b * 2 + 1) * NWORDS + wd];
    float s = 0.f;
    for (int t = start; t < end; ++t)
        s += (float)x[((size_t)b * TFRAMES + t) * HCH + ci];
    float cnt = fmaxf((float)(end - start), 1.f);
    float m = s / cnt;
    if (wd >= lens[b]) m = 0.f;
    out[((size_t)b * NWORDS + wd) * HCH + ci] = (__bf16)m;
}

// ---------------------------------------------------------------------------
// Final 128 -> 1 conv (tiny): x bf16 [B][W][128] -> out f32 [B][1][W]
// ---------------------------------------------------------------------------
__global__ void final_conv_kernel(const __bf16* __restrict__ x, const float* __restrict__ w,
                                  const float* __restrict__ bias, float* __restrict__ out) {
    int i = blockIdx.x * blockDim.x + threadIdx.x;
    if (i >= NBATCH * NWORDS) return;
    int b = i / NWORDS, wd = i % NWORDS;
    float s = bias[0];
    for (int k = 0; k < 5; ++k) {
        int t = wd + k - 2;
        if (t < 0 || t >= NWORDS) continue;
        const __bf16* row = x + ((size_t)b * NWORDS + t) * HCH;
        for (int ci = 0; ci < HCH; ++ci) s += w[ci * 5 + k] * (float)row[ci];
    }
    out[i] = s;
}

// ---------------------------------------------------------------------------
// Launcher
// Inputs: 0 features | 1-4 enc_ws | 5-8 enc_bs | 9-12 dec_ws | 13-16 dec_bs
//         17 word_bounds | 18 word_lengths
// ---------------------------------------------------------------------------
extern "C" void kernel_launch(void* const* d_in, const int* in_sizes, int n_in,
                              void* d_out, int out_size, void* d_ws, size_t ws_size,
                              hipStream_t stream) {
    const float* features = (const float*)d_in[0];
    const float* ew[4] = {(const float*)d_in[1], (const float*)d_in[2],
                          (const float*)d_in[3], (const float*)d_in[4]};
    const float* eb[4] = {(const float*)d_in[5], (const float*)d_in[6],
                          (const float*)d_in[7], (const float*)d_in[8]};
    const float* dw[4] = {(const float*)d_in[9], (const float*)d_in[10],
                          (const float*)d_in[11], (const float*)d_in[12]};
    const float* db[4] = {(const float*)d_in[13], (const float*)d_in[14],
                          (const float*)d_in[15], (const float*)d_in[16]};
    const int* bounds = (const int*)d_in[17];
    const int* lens   = (const int*)d_in[18];
    float* out = (float*)d_out;

    char* ws = (char*)d_ws;
    size_t off = 0;
    auto take = [&](size_t bytes) -> char* {
        char* p = ws + off;
        off += (bytes + 255) & ~(size_t)255;
        return p;
    };

    const size_t actBytes = (size_t)NBATCH * TFRAMES * HCH * sizeof(__bf16);  // 16.8 MB
    __bf16* xA = (__bf16*)take(actBytes);
    __bf16* xB = (__bf16*)take(actBytes);   // also holds the 96-wide transposed input
    __bf16* pA = (__bf16*)take((size_t)NBATCH * NWORDS * HCH * sizeof(__bf16));
    __bf16* pB = (__bf16*)take((size_t)NBATCH * NWORDS * HCH * sizeof(__bf16));
    __bf16* wp[7];
    const int nch96 = 96 * 5 / 32;    // 15
    const int nch128 = 128 * 5 / 32;  // 20
    wp[0] = (__bf16*)take((size_t)8 * nch96 * 512 * sizeof(__bf16));
    for (int i = 1; i < 7; ++i)
        wp[i] = (__bf16*)take((size_t)8 * nch128 * 512 * sizeof(__bf16));

    // ---- prep: transpose input + pack all conv weights into A-fragment order
    {
        size_t total = (size_t)NBATCH * TFRAMES * 96;
        transpose_in_kernel<<<dim3((unsigned)((total + 255) / 256)), 256, 0, stream>>>(features, xB);
    }
    pack_weights_kernel<<<(8 * nch96 * 512 + 255) / 256, 256, 0, stream>>>(ew[0], wp[0], 80, 96);
    for (int l = 1; l < 4; ++l)
        pack_weights_kernel<<<(8 * nch128 * 512 + 255) / 256, 256, 0, stream>>>(ew[l], wp[l], 128, 128);
    for (int l = 0; l < 3; ++l)
        pack_weights_kernel<<<(8 * nch128 * 512 + 255) / 256, 256, 0, stream>>>(dw[l], wp[4 + l], 128, 128);

    // ---- encoder: Conv-ReLU x3 + Conv (T = 4096)
    dim3 gEnc(TFRAMES / 64, NBATCH);
    conv_wmma_kernel<96, true><<<gEnc, 256, 0, stream>>>(xB, wp[0], eb[0], xA, TFRAMES);
    conv_wmma_kernel<128, true><<<gEnc, 256, 0, stream>>>(xA, wp[1], eb[1], xB, TFRAMES);
    conv_wmma_kernel<128, true><<<gEnc, 256, 0, stream>>>(xB, wp[2], eb[2], xA, TFRAMES);
    conv_wmma_kernel<128, false><<<gEnc, 256, 0, stream>>>(xA, wp[3], eb[3], xB, TFRAMES);

    // ---- segment mean pool -> [B][W][128]
    pool_kernel<<<dim3(NWORDS, NBATCH), 128, 0, stream>>>(xB, bounds, lens, pA);

    // ---- decoder: Conv-ReLU x3 (T = 256) + scalar 128->1 head
    dim3 gDec(NWORDS / 64, NBATCH);
    conv_wmma_kernel<128, true><<<gDec, 256, 0, stream>>>(pA, wp[4], db[0], pB, NWORDS);
    conv_wmma_kernel<128, true><<<gDec, 256, 0, stream>>>(pB, wp[5], db[1], pA, NWORDS);
    conv_wmma_kernel<128, true><<<gDec, 256, 0, stream>>>(pA, wp[6], db[2], pB, NWORDS);
    final_conv_kernel<<<(NBATCH * NWORDS + 255) / 256, 256, 0, stream>>>(pB, dw[3], db[3], out);
}
